// EncoderLstm_one_fc_24008867184782
// MI455X (gfx1250) — matrix operand order repaired and checked by reading, not compiled, not run on hardware
//
#include <hip/hip_runtime.h>
#include <math.h>

// ---------------------------------------------------------------------------
// EncoderLstm for MI455X (gfx1250, wave32, WMMA).
//   All GEMMs: v_wmma_f32_16x16x32_bf16, f32 accumulate. Operands are
//   pre-converted to bf16 once (feats/weights/emb/h), so GEMM inner loops are
//   pure b128 loads + WMMA. Each wave register-blocks a 32x32 output tile
//   (2x2 WMMA tiles): 4 WMMAs per 8 b128 loads per 32-deep k-step.
//   Working set (~66MB ws + 90MB inputs) is L2-resident (192MB).
// ---------------------------------------------------------------------------

typedef __attribute__((ext_vector_type(16))) __bf16 v16bf;
typedef __attribute__((ext_vector_type(8)))  __bf16 v8bf;
typedef __attribute__((ext_vector_type(8)))  float  v8f;

#define B_   128
#define T_   32
#define FIN  2048
#define H_   1024
#define BT   (B_ * T_)      // 4096
#define G4   (4 * H_)       // 4096
#define BN_EPS 1e-5f

struct FragBytes { float4 lo; float4 hi; };   // 32 bytes == v16bf

// Two 16-byte bf16 runs -> one v16bf fragment (elements 0-7 = run0, 8-15 = run1).
__device__ __forceinline__ v16bf ld_frag(const __bf16* p0, const __bf16* p1) {
  FragBytes fb;
  fb.lo = *(const float4*)p0;
  fb.hi = *(const float4*)p1;
  return __builtin_bit_cast(v16bf, fb);
}

// A fragment (16x32, rows m, K-slice k0..k0+31), ISA 7.12.2 layout:
//   lanes 0-15: row m, K = {k0..k0+7, k0+16..k0+23}
//   lanes16-31: row m, K = {k0+8..k0+15, k0+24..k0+31}
// pa = A + (m0+lr)*lda + half*8  (then advance by k)
#define LD_A(pa) ld_frag((pa), (pa) + 16)
// B fragment (32x16 = W^T tile): lane n holds row (n0+n) of W;
//   same K order as A: lanes 0-15 K={k0..k0+7, k0+16..k0+23}, lanes 16-31 K={k0+8..k0+15, k0+24..k0+31}.
// pb = W + (n0+lr)*ldk + half*8
#define LD_B(pb) ld_frag((pb), (pb) + 16)

#define WMMA_BF16(a, b, c) \
  __builtin_amdgcn_wmma_f32_16x16x32_bf16(false, (a), false, (b), (short)0, (c), false, false)

// ---- generic C = A @ W^T (+bias), 32x32 output per wave --------------------
__global__ void wmma_gemm_bias(const __bf16* __restrict__ A, int lda,
                               const __bf16* __restrict__ W, int ldk,
                               const float* __restrict__ bias,
                               float* __restrict__ C, int ldc,
                               int wn_tiles, int K) {
  const int wave = threadIdx.x >> 5;
  const int lane = threadIdx.x & 31;
  const int lr = lane & 15, half = lane >> 4;
  const int w = blockIdx.x * 8 + wave;
  const int m0 = (w / wn_tiles) * 32;
  const int n0 = (w % wn_tiles) * 32;

  const __bf16* pa0 = A + (size_t)(m0 + lr) * lda + half * 8;
  const __bf16* pa1 = pa0 + (size_t)16 * lda;
  const __bf16* pb0 = W + (size_t)(n0 + lr) * ldk + half * 8;
  const __bf16* pb1 = pb0 + (size_t)16 * ldk;

  v8f acc00 = {}, acc01 = {}, acc10 = {}, acc11 = {};
  for (int k0 = 0; k0 < K; k0 += 32) {
    v16bf a0 = LD_A(pa0), a1 = LD_A(pa1);
    v16bf b0 = LD_B(pb0), b1 = LD_B(pb1);
    pa0 += 32; pa1 += 32; pb0 += 32; pb1 += 32;
    acc00 = WMMA_BF16(a0, b0, acc00);
    acc01 = WMMA_BF16(a0, b1, acc01);
    acc10 = WMMA_BF16(a1, b0, acc10);
    acc11 = WMMA_BF16(a1, b1, acc11);
  }
  const float bv0 = bias[n0 + lr], bv1 = bias[n0 + 16 + lr];
#pragma unroll
  for (int r = 0; r < 8; ++r) {
    const size_t row0 = (size_t)(m0 + r + 8 * half) * ldc;
    const size_t row1 = (size_t)(m0 + 16 + r + 8 * half) * ldc;
    C[row0 + n0 + lr]      = acc00[r] + bv0;
    C[row0 + n0 + 16 + lr] = acc01[r] + bv1;
    C[row1 + n0 + lr]      = acc10[r] + bv0;
    C[row1 + n0 + 16 + lr] = acc11[r] + bv1;
  }
}

// ---- per-step gate GEMM: g = emb_t @ W_ih^T + h @ W_hh^T (32x32/wave) ------
__global__ void lstm_step_gemm(const __bf16* __restrict__ emb,
                               const __bf16* __restrict__ h,
                               const __bf16* __restrict__ Wih,
                               const __bf16* __restrict__ Whh,
                               float* __restrict__ g, int t) {
  const int wave = threadIdx.x >> 5;
  const int lane = threadIdx.x & 31;
  const int lr = lane & 15, half = lane >> 4;
  const int w = blockIdx.x * 8 + wave;
  const int wn_tiles = G4 / 32;                    // 128
  const int m0 = (w / wn_tiles) * 32;
  const int n0 = (w % wn_tiles) * 32;

  v8f acc00 = {}, acc01 = {}, acc10 = {}, acc11 = {};

  // phase 1: emb_t rows (stride T_*H_) vs W_ih
  {
    const __bf16* pa0 = emb + (size_t)t * H_ + (size_t)(m0 + lr) * (T_ * H_) + half * 8;
    const __bf16* pa1 = pa0 + (size_t)16 * (T_ * H_);
    const __bf16* pb0 = Wih + (size_t)(n0 + lr) * H_ + half * 8;
    const __bf16* pb1 = pb0 + (size_t)16 * H_;
    for (int k0 = 0; k0 < H_; k0 += 32) {
      v16bf a0 = LD_A(pa0), a1 = LD_A(pa1);
      v16bf b0 = LD_B(pb0), b1 = LD_B(pb1);
      pa0 += 32; pa1 += 32; pb0 += 32; pb1 += 32;
      acc00 = WMMA_BF16(a0, b0, acc00);
      acc01 = WMMA_BF16(a0, b1, acc01);
      acc10 = WMMA_BF16(a1, b0, acc10);
      acc11 = WMMA_BF16(a1, b1, acc11);
    }
  }
  // phase 2: h rows (stride H_) vs W_hh
  {
    const __bf16* pa0 = h + (size_t)(m0 + lr) * H_ + half * 8;
    const __bf16* pa1 = pa0 + (size_t)16 * H_;
    const __bf16* pb0 = Whh + (size_t)(n0 + lr) * H_ + half * 8;
    const __bf16* pb1 = pb0 + (size_t)16 * H_;
    for (int k0 = 0; k0 < H_; k0 += 32) {
      v16bf a0 = LD_A(pa0), a1 = LD_A(pa1);
      v16bf b0 = LD_B(pb0), b1 = LD_B(pb1);
      pa0 += 32; pa1 += 32; pb0 += 32; pb1 += 32;
      acc00 = WMMA_BF16(a0, b0, acc00);
      acc01 = WMMA_BF16(a0, b1, acc01);
      acc10 = WMMA_BF16(a1, b0, acc10);
      acc11 = WMMA_BF16(a1, b1, acc11);
    }
  }
#pragma unroll
  for (int r = 0; r < 8; ++r) {
    const size_t row0 = (size_t)(m0 + r + 8 * half) * G4;
    const size_t row1 = (size_t)(m0 + 16 + r + 8 * half) * G4;
    g[row0 + n0 + lr]      = acc00[r];
    g[row0 + n0 + 16 + lr] = acc01[r];
    g[row1 + n0 + lr]      = acc10[r];
    g[row1 + n0 + 16 + lr] = acc11[r];
  }
}

// ---- one-time f32 -> bf16 conversion (8 elements / thread, b128 traffic) ---
__global__ void cvt_f32_bf16_x8(const float* __restrict__ in,
                                __bf16* __restrict__ out) {
  const size_t base = ((size_t)blockIdx.x * blockDim.x + threadIdx.x) * 8;
  float4 x = *(const float4*)(in + base);
  float4 y = *(const float4*)(in + base + 4);
  v8bf v;
  v[0] = (__bf16)x.x; v[1] = (__bf16)x.y; v[2] = (__bf16)x.z; v[3] = (__bf16)x.w;
  v[4] = (__bf16)y.x; v[5] = (__bf16)y.y; v[6] = (__bf16)y.z; v[7] = (__bf16)y.w;
  *(v8bf*)(out + base) = v;
}

// ---- BatchNorm stats over B*T rows (per-column mean / rsqrt(var+eps)) ------
__global__ void bn_stats(const float* __restrict__ Y,
                         float* __restrict__ mean, float* __restrict__ rstd) {
  const int col = blockIdx.x * blockDim.x + threadIdx.x;   // 1024 threads
  float s = 0.0f, s2 = 0.0f;
  for (int r = 0; r < BT; ++r) {
    float v = Y[(size_t)r * H_ + col];
    s += v; s2 += v * v;
  }
  const float m = s * (1.0f / BT);
  const float var = s2 * (1.0f / BT) - m * m;
  mean[col] = m;
  rstd[col] = rsqrtf(var + BN_EPS);
}

// ---- BN apply + ReLU + mask -> bf16 emb ------------------------------------
__global__ void bn_apply_relu_mask(const float* __restrict__ Y,
                                   const float* __restrict__ mean,
                                   const float* __restrict__ rstd,
                                   const float* __restrict__ gamma,
                                   const float* __restrict__ beta,
                                   const float* __restrict__ mask,
                                   __bf16* __restrict__ emb) {
  const size_t idx = (size_t)blockIdx.x * blockDim.x + threadIdx.x;
  const int row = (int)(idx >> 10);       // H_ = 1024
  const int col = (int)(idx & (H_ - 1));
  float v = Y[idx];
  v = (v - mean[col]) * rstd[col] * gamma[col] + beta[col];
  emb[idx] = (__bf16)(fmaxf(v, 0.0f) * mask[row]);
}

__device__ __forceinline__ float sig(float x) { return 1.0f / (1.0f + __expf(-x)); }

// ---- LSTM gates / state update for one timestep ----------------------------
__global__ void lstm_gates(const float* __restrict__ g,
                           const float* __restrict__ bih,
                           const float* __restrict__ bhh,
                           const float* __restrict__ mask,
                           __bf16* __restrict__ h, float* __restrict__ c,
                           float* __restrict__ out, int t) {
  const size_t idx = (size_t)blockIdx.x * blockDim.x + threadIdx.x; // B_*H_
  const int b  = (int)(idx >> 10);
  const int hc = (int)(idx & (H_ - 1));
  const float* gr = g + (size_t)b * G4;
  const float gi = gr[hc]          + bih[hc]          + bhh[hc];
  const float gf = gr[H_ + hc]     + bih[H_ + hc]     + bhh[H_ + hc];
  const float gg = gr[2 * H_ + hc] + bih[2 * H_ + hc] + bhh[2 * H_ + hc];
  const float go = gr[3 * H_ + hc] + bih[3 * H_ + hc] + bhh[3 * H_ + hc];
  const float i = sig(gi), f = sig(gf), gt = tanhf(gg), o = sig(go);
  const float m = mask[b * T_ + t];
  const float cc = f * c[idx] + i * gt;      // pre-mask cell (ref semantics)
  const float hn = o * tanhf(cc) * m;
  c[idx] = cc * m;
  h[idx] = (__bf16)hn;                       // bf16 h feeds next-step GEMM
  out[(size_t)(b * T_ + t) * H_ + hc] = hn;
}

__global__ void zero_u32(unsigned int* __restrict__ p, int n) {
  const int i = blockIdx.x * blockDim.x + threadIdx.x;
  if (i < n) p[i] = 0u;
}

extern "C" void kernel_launch(void* const* d_in, const int* in_sizes, int n_in,
                              void* d_out, int out_size, void* d_ws, size_t ws_size,
                              hipStream_t stream) {
  (void)in_sizes; (void)n_in; (void)out_size; (void)ws_size;
  const float* feats = (const float*)d_in[0];   // [B,T,FIN]
  const float* fmask = (const float*)d_in[1];   // [B,T]
  const float* W_emb = (const float*)d_in[2];   // [H,FIN]
  const float* b_emb = (const float*)d_in[3];   // [H]
  const float* gamma = (const float*)d_in[4];
  const float* beta  = (const float*)d_in[5];
  const float* W_ih  = (const float*)d_in[6];   // [4H,H]
  const float* b_ih  = (const float*)d_in[7];
  const float* W_hh  = (const float*)d_in[8];   // [4H,H]
  const float* b_hh  = (const float*)d_in[9];
  float* out = (float*)d_out;                   // [B,T,H]

  // workspace layout (bytes), all 256-aligned; ~66 MB total
  char* w = (char*)d_ws;
  __bf16* feats_bf = (__bf16*)w;                 w += (size_t)BT * FIN * 2;   // 16MB
  __bf16* Wemb_bf  = (__bf16*)w;                 w += (size_t)H_ * FIN * 2;   //  4MB
  __bf16* Wih_bf   = (__bf16*)w;                 w += (size_t)G4 * H_ * 2;    //  8MB
  __bf16* Whh_bf   = (__bf16*)w;                 w += (size_t)G4 * H_ * 2;    //  8MB
  float*  Y        = (float*)w;                  w += (size_t)BT * H_ * 4;    // 16MB
  __bf16* emb_bf   = (__bf16*)w;                 w += (size_t)BT * H_ * 2;    //  8MB
  float*  mean     = (float*)w;                  w += H_ * 4;
  float*  rstd     = (float*)w;                  w += H_ * 4;
  __bf16* h_bf     = (__bf16*)w;                 w += (size_t)B_ * H_ * 2;
  float*  c        = (float*)w;                  w += (size_t)B_ * H_ * 4;
  float*  g        = (float*)w;                  w += (size_t)B_ * G4 * 4;

  // 0) one-time f32->bf16 conversions (pure vmem + v_cvt_pk, b128 traffic)
  cvt_f32_bf16_x8<<<(size_t)BT * FIN / 8 / 256, 256, 0, stream>>>(feats, feats_bf);
  cvt_f32_bf16_x8<<<(size_t)H_ * FIN / 8 / 256, 256, 0, stream>>>(W_emb, Wemb_bf);
  cvt_f32_bf16_x8<<<(size_t)G4 * H_ / 8 / 256, 256, 0, stream>>>(W_ih, Wih_bf);
  cvt_f32_bf16_x8<<<(size_t)G4 * H_ / 8 / 256, 256, 0, stream>>>(W_hh, Whh_bf);

  // 1) h0 = c0 = 0   (h_bf: B*H bf16 = B*H/2 u32; c: B*H f32)
  zero_u32<<<(B_ * H_ / 2 + 255) / 256, 256, 0, stream>>>((unsigned int*)h_bf, B_ * H_ / 2);
  zero_u32<<<(B_ * H_ + 255) / 256, 256, 0, stream>>>((unsigned int*)c, B_ * H_);

  // 2) Y = feats @ W_emb^T + b_emb   (4096x1024, K=2048), 32x32 per wave
  {
    const int wn = H_ / 32;                      // 32
    const int waves = (BT / 32) * wn;            // 4096
    wmma_gemm_bias<<<waves / 8, 256, 0, stream>>>(
        feats_bf, FIN, Wemb_bf, FIN, b_emb, Y, H_, wn, FIN);
  }

  // 3) BN stats + apply(ReLU, mask) -> bf16 emb
  bn_stats<<<H_ / 256, 256, 0, stream>>>(Y, mean, rstd);
  bn_apply_relu_mask<<<(size_t)BT * H_ / 256, 256, 0, stream>>>(
      Y, mean, rstd, gamma, beta, fmask, emb_bf);

  // 4) LSTM scan: per step, fused [emb_t | h] @ [W_ih | W_hh]^T then gates
  for (int t = 0; t < T_; ++t) {
    const int waves = (B_ / 32) * (G4 / 32);     // 512
    lstm_step_gemm<<<waves / 8, 256, 0, stream>>>(emb_bf, h_bf, Wih_bf, Whh_bf, g, t);
    lstm_gates<<<B_ * H_ / 256, 256, 0, stream>>>(g, b_ih, b_hh, fmask, h_bf, c, out, t);
  }
}
